// Wav2vecGumbelVectorQuantizer_24988119728779
// MI455X (gfx1250) — matrix operand order repaired and checked by reading, not compiled, not run on hardware
//
#include <hip/hip_runtime.h>
#include <hip/hip_bf16.h>

// ---------------- problem constants ----------------
#define BB   8
#define TT   512
#define FF   256
#define GG   2
#define VV   8192
#define DG   8
#define BT   (BB*TT)          // 4096
#define NN   (GG*VV)          // 16384  (GEMM N)
#define ROWS (BT*GG)          // 8192   (softmax rows)

typedef __bf16 bf16;
typedef __attribute__((ext_vector_type(16))) __bf16 v16bf;
typedef __attribute__((ext_vector_type(8)))  __bf16 v8bf;
typedef __attribute__((ext_vector_type(8)))  float  v8f;
typedef __attribute__((ext_vector_type(4)))  int    v4i;

// ---------------- optional CDNA5 async global->LDS path ----------------
#if defined(__has_builtin)
#  if __has_builtin(__builtin_amdgcn_global_load_async_to_lds_b128)
#    define USE_ASYNC_LDS 1
#  endif
#endif
#ifndef USE_ASYNC_LDS
#  define USE_ASYNC_LDS 0
#endif

#define AS1 __attribute__((address_space(1)))
#define AS3 __attribute__((address_space(3)))

__device__ inline void async_copy16(const bf16* gsrc, bf16* ldst) {
#if USE_ASYNC_LDS
    __builtin_amdgcn_global_load_async_to_lds_b128(
        (AS1 v4i*)gsrc, (AS3 v4i*)ldst, 0, 0);
#else
    *(uint4*)ldst = *(const uint4*)gsrc;
#endif
}
__device__ inline void async_wait0() {
#if USE_ASYNC_LDS
#  if __has_builtin(__builtin_amdgcn_s_wait_asynccnt)
    __builtin_amdgcn_s_wait_asynccnt(0);
#  else
    asm volatile("s_wait_asynccnt 0" ::: "memory");
#  endif
#endif
}

// ---------------- reduction helpers (wave32) ----------------
__device__ inline float wave_max(float v) {
#pragma unroll
    for (int off = 16; off; off >>= 1) v = fmaxf(v, __shfl_xor(v, off, 32));
    return v;
}
__device__ inline float wave_sum(float v) {
#pragma unroll
    for (int off = 16; off; off >>= 1) v += __shfl_xor(v, off, 32);
    return v;
}
// 256-thread (8-wave) block reductions; sbuf must be >= 8 floats.
__device__ inline float block_max(float v, float* sbuf) {
    int lane = threadIdx.x & 31, wv = threadIdx.x >> 5;
    v = wave_max(v);
    __syncthreads();
    if (lane == 0) sbuf[wv] = v;
    __syncthreads();
    float r = sbuf[0];
#pragma unroll
    for (int i = 1; i < 8; ++i) r = fmaxf(r, sbuf[i]);
    return r;
}
__device__ inline float block_sum(float v, float* sbuf) {
    int lane = threadIdx.x & 31, wv = threadIdx.x >> 5;
    v = wave_sum(v);
    __syncthreads();
    if (lane == 0) sbuf[wv] = v;
    __syncthreads();
    float r = 0.f;
#pragma unroll
    for (int i = 0; i < 8; ++i) r += sbuf[i];
    return r;
}

// ---------------- kernel 1: fp32 -> bf16 conversion ----------------
__global__ void __launch_bounds__(256)
convert_bf16_kernel(const float* __restrict__ x, const float* __restrict__ w,
                    bf16* __restrict__ xh, bf16* __restrict__ wh) {
    int i = blockIdx.x * 256 + threadIdx.x;
    int stride = gridDim.x * 256;
    for (int j = i; j < BT * FF; j += stride) xh[j] = (bf16)x[j];
    for (int j = i; j < NN * FF; j += stride) wh[j] = (bf16)w[j];
}

// ---------------- kernel 2: hidden = x @ W^T + b  (bf16 WMMA) ----------------
// block = 256 threads = 8 waves (4 M x 2 N); block tile = 64(M) x 128(N)
// wave tile = 16 x 64 (4 accumulators); K = 256 in 8 steps of 32.
// A: LDS-resident for the whole K (padded rows). B: double-buffered LDS tile
// staged per K-step via async global->LDS (4x M-reuse of every B element).
#define A_STRIDE 264     // 256 + 8 bf16 pad  -> conflict-free b128 fragment reads
#define B_STRIDE 40      // 32 + 8 bf16 pad
__global__ void __launch_bounds__(256)
gemm_hidden_kernel(const bf16* __restrict__ xh, const bf16* __restrict__ wh,
                   const float* __restrict__ bproj, float* __restrict__ h) {
    __shared__ __align__(16) bf16 Alds[64 * A_STRIDE];        // ~33 KB
    __shared__ __align__(16) bf16 Blds[2][128 * B_STRIDE];    // ~20 KB

    const int tid = threadIdx.x;
    const int m_base = blockIdx.y * 64;
    const size_t n_base = (size_t)blockIdx.x * 128;

    // ---- stage the full 64x256 A tile (one time) ----
#pragma unroll
    for (int j = 0; j < 8; ++j) {
        int uu = tid + j * 256;               // 2048 16B-units
        int row = uu >> 5, seg = uu & 31;
        *(uint4*)&Alds[row * A_STRIDE + seg * 8] =
            *(const uint4*)(xh + (size_t)(m_base + row) * FF + seg * 8);
    }
    // ---- stage B tile for kb=0 into buffer 0 ----
    const bf16* wblk = wh + n_base * FF;      // [128 rows][256 K]
#pragma unroll
    for (int j = 0; j < 2; ++j) {
        int uu = tid + j * 256;               // 512 16B-units (128 rows x 4 segs)
        int row = uu >> 2, seg = uu & 3;
        async_copy16(wblk + (size_t)row * FF + seg * 8,
                     &Blds[0][row * B_STRIDE + seg * 8]);
    }
    async_wait0();
    __syncthreads();

    const int lane = tid & 31, wv = tid >> 5;
    const int wm = wv >> 1, wn = wv & 1;
    const int half = lane >> 4;               // lane group 0/1
    const int mrow = lane & 15;               // A row / B-C column within tile
    const int arow = (wm * 16 + mrow) * A_STRIDE;

    v8f acc[4] = {};

#pragma unroll
    for (int ks = 0; ks < 8; ++ks) {
        const int kb = ks * 32;
        const int cur = ks & 1;

        // prefetch next K-step's B tile into the other buffer (overlaps WMMA)
        if (ks < 7) {
#pragma unroll
            for (int j = 0; j < 2; ++j) {
                int uu = tid + j * 256;
                int row = uu >> 2, seg = uu & 3;
                async_copy16(wblk + (size_t)row * FF + (kb + 32) + seg * 8,
                             &Blds[cur ^ 1][row * B_STRIDE + seg * 8]);
            }
        }

        // A fragment (16x32 bf16): two contiguous 8-elem chunks per lane
        v8bf alo = *(const v8bf*)&Alds[arow + kb + half * 8];
        v8bf ahi = *(const v8bf*)&Alds[arow + kb + 16 + half * 8];
        v16bf a = __builtin_shufflevector(alo, ahi,
                    0,1,2,3,4,5,6,7,8,9,10,11,12,13,14,15);
#pragma unroll
        for (int t = 0; t < 4; ++t) {
            // B fragment (32x16 bf16): lane holds col N=lane%16, K chunks half*8 and 16+half*8
            const bf16* bp = &Blds[cur][(wn * 64 + t * 16 + mrow) * B_STRIDE + half * 8];
            v8bf blo = *(const v8bf*)(bp);
            v8bf bhi = *(const v8bf*)(bp + 16);
            v16bf b = __builtin_shufflevector(blo, bhi,
                        0,1,2,3,4,5,6,7,8,9,10,11,12,13,14,15);
            acc[t] = __builtin_amdgcn_wmma_f32_16x16x32_bf16(
                        false, a, false, b, (short)0, acc[t], false, false);
        }
        async_wait0();
        __syncthreads();
    }

#pragma unroll
    for (int t = 0; t < 4; ++t) {
        size_t n = n_base + wn * 64 + t * 16 + mrow;   // C: N = lane%16
        float bb = bproj[n];
#pragma unroll
        for (int r = 0; r < 8; ++r) {                  // C: M = (lane/16)*8 + r
            int m = m_base + wm * 16 + half * 8 + r;
            h[(size_t)m * NN + n] = acc[t][r] + bb;
        }
    }
}

// ---------------- kernel 3: fused gumbel softmax / argmax / codevector GEMV ----
// one block per (bt, g) row of [ROWS, V]; 256 threads, 32 columns per thread
__global__ void __launch_bounds__(256)
softmax_cv_kernel(const float* __restrict__ h, const float* __restrict__ u,
                  const float* __restrict__ emb, float* __restrict__ out,
                  float* __restrict__ rmax, float* __restrict__ rinv) {
    __shared__ float sbuf[8];
    __shared__ float avals[8];
    __shared__ int   aidx[8];
    __shared__ float cvred[8][8];

    const int r = blockIdx.x;
    const int bt = r >> 1, g = r & 1;
    const int tid = threadIdx.x, lane = tid & 31, wv = tid >> 5;
    const size_t rowoff = (size_t)r * VV;

    float noisy[32];
    float mn = -INFINITY, mc = -INFINITY, bv = -INFINITY;
    int bi = 0;
#pragma unroll
    for (int i = 0; i < 32; ++i) {
        int v = i * 256 + tid;
        float hv = h[rowoff + v];
        float uu = u[rowoff + v];
        uu = fminf(fmaxf(uu, 1.1754944e-38f), 1.0f);
        float nz = hv - __logf(-__logf(uu));   // h + gumbel, temperature = 1
        noisy[i] = nz;
        mc = fmaxf(mc, hv);
        mn = fmaxf(mn, nz);
        if (nz > bv) { bv = nz; bi = v; }
    }
    float Mn = block_max(mn, sbuf);
    float Mc = block_max(mc, sbuf);

    // block argmax of noisy (ties -> lowest index)
#pragma unroll
    for (int off = 16; off; off >>= 1) {
        float ov = __shfl_xor(bv, off, 32);
        int   oi = __shfl_xor(bi, off, 32);
        if (ov > bv || (ov == bv && oi < bi)) { bv = ov; bi = oi; }
    }
    __syncthreads();
    if (lane == 0) { avals[wv] = bv; aidx[wv] = bi; }
    __syncthreads();
    float BV = avals[0]; int BI = aidx[0];
#pragma unroll
    for (int i = 1; i < 8; ++i)
        if (avals[i] > BV || (avals[i] == BV && aidx[i] < BI)) { BV = avals[i]; BI = aidx[i]; }

    float sn = 0.f, sc = 0.f;
#pragma unroll
    for (int i = 0; i < 32; ++i) {
        int v = i * 256 + tid;
        sn += __expf(noisy[i] - Mn);
        sc += __expf(h[rowoff + v] - Mc);      // clean softmax stats (L2 hit)
    }
    float Sn = block_sum(sn, sbuf);
    float Sc = block_sum(sc, sbuf);
    float inv_sn = 1.0f / Sn;

    // codevectors: probs[V] @ emb[g][V][8], accumulated per thread then reduced
    float cv[8] = {0.f, 0.f, 0.f, 0.f, 0.f, 0.f, 0.f, 0.f};
#pragma unroll
    for (int i = 0; i < 32; ++i) {
        int v = i * 256 + tid;
        float p = __expf(noisy[i] - Mn) * inv_sn;
        const float4* e4 = (const float4*)(emb + ((size_t)g * VV + v) * DG);
        float4 e0 = e4[0], e1 = e4[1];
        cv[0] += p * e0.x; cv[1] += p * e0.y; cv[2] += p * e0.z; cv[3] += p * e0.w;
        cv[4] += p * e1.x; cv[5] += p * e1.y; cv[6] += p * e1.z; cv[7] += p * e1.w;
    }
#pragma unroll
    for (int d = 0; d < 8; ++d) {
        float s = wave_sum(cv[d]);
        if (lane == 0) cvred[wv][d] = s;
    }
    __syncthreads();
    if (tid < 8) {
        float s = 0.f;
#pragma unroll
        for (int w = 0; w < 8; ++w) s += cvred[w][tid];
        out[(size_t)bt * (GG * DG) + g * DG + tid] = s;   // codevectors
    }
    if (tid == 0) {
        out[BT * GG * DG + 1 + (size_t)bt * GG + g] = (float)BI;  // targets_idx
        rmax[r] = Mc;
        rinv[r] = 1.0f / Sc;
    }
}

// ---------------- kernel 4: marginal over masked rows, column reduction -------
// 64 blocks x 256 threads = one thread per (g, v) column
__global__ void __launch_bounds__(256)
marginal_kernel(const float* __restrict__ h, const int* __restrict__ mask,
                const float* __restrict__ rmax, const float* __restrict__ rinv,
                float* __restrict__ marg) {
    __shared__ float sbuf[8];
    const int gid = blockIdx.x * 256 + threadIdx.x;   // 0..NN-1
    const int g = gid >> 13;                          // / VV
    const int v = gid & (VV - 1);

    int mloc = 0;
#pragma unroll
    for (int j = 0; j < BT / 256; ++j) mloc += mask[threadIdx.x + j * 256];
    float mcnt = block_sum((float)mloc, sbuf);

    float s = 0.f;
    for (int bt = 0; bt < BT; ++bt) {
        if (mask[bt]) {
            int row = bt * GG + g;
            s += __expf(h[(size_t)row * VV + v] - rmax[row]) * rinv[row];
        }
    }
    marg[gid] = s / mcnt;
}

// ---------------- kernel 5: entropy -> perplexity scalar ----------------------
__global__ void __launch_bounds__(256)
perplexity_kernel(const float* __restrict__ marg, float* __restrict__ out) {
    __shared__ float sbuf[8];
    float px = 0.f;
    for (int g = 0; g < GG; ++g) {
        float e = 0.f;
#pragma unroll
        for (int j = 0; j < VV / 256; ++j) {
            float m = marg[g * VV + threadIdx.x + j * 256];
            e += -m * __logf(m + 1e-7f);
        }
        e = block_sum(e, sbuf);
        px += __expf(e);
    }
    if (threadIdx.x == 0) out[BT * GG * DG] = px;   // perplexity at offset 65536
}

// ---------------- launcher ----------------------------------------------------
extern "C" void kernel_launch(void* const* d_in, const int* in_sizes, int n_in,
                              void* d_out, int out_size, void* d_ws, size_t ws_size,
                              hipStream_t stream) {
    const float* x     = (const float*)d_in[0];   // [B,T,F]
    const float* u     = (const float*)d_in[1];   // [B*T*G, V]
    const float* emb   = (const float*)d_in[2];   // [1, G*V, DG]
    const float* W     = (const float*)d_in[3];   // [G*V, F]
    const float* bproj = (const float*)d_in[4];   // [G*V]
    const int*   mask  = (const int*)d_in[5];     // [B,T]
    float* out = (float*)d_out;

    char* ws = (char*)d_ws;
    bf16*  xh   = (bf16*)ws;                                  // 2 MB
    bf16*  wh   = (bf16*)(ws + ((size_t)2  << 20));           // 8 MB
    float* h    = (float*)(ws + ((size_t)10 << 20));          // 256 MB
    float* rmax = (float*)(ws + ((size_t)266 << 20));         // 32 KB
    float* rinv = rmax + ROWS;                                // 32 KB
    float* marg = rinv + ROWS;                                // 64 KB

    convert_bf16_kernel<<<2048, 256, 0, stream>>>(x, W, xh, wh);
    gemm_hidden_kernel<<<dim3(NN / 128, BT / 64), 256, 0, stream>>>(xh, wh, bproj, h);
    softmax_cv_kernel<<<ROWS, 256, 0, stream>>>(h, u, emb, out, rmax, rinv);
    marginal_kernel<<<NN / 256, 256, 0, stream>>>(h, mask, rmax, rinv, marg);
    perplexity_kernel<<<1, 256, 0, stream>>>(marg, out);
}
